// UltimateOptimizationPiMoE_44942537785681
// MI455X (gfx1250) — matrix operand order repaired and checked by reading, not compiled, not run on hardware
//
#include <hip/hip_runtime.h>
#include <math.h>

// Problem constants (from reference)
#define D_   1024
#define B_   8
#define S_   256
#define E_   4
#define H_   16
#define DH_  64
#define TOK  (B_*S_)

typedef __attribute__((ext_vector_type(16))) _Float16 v16h;
typedef __attribute__((ext_vector_type(8)))  float    v8f;
typedef __attribute__((ext_vector_type(4)))  float    f4;
typedef __attribute__((ext_vector_type(4)))  int      v4i;

// ---------- CDNA5 async-to-LDS (guarded; falls back to plain LDS stores) ----------
#if defined(__gfx1250__) && __has_builtin(__builtin_amdgcn_global_load_async_to_lds_b128) && __has_builtin(__builtin_amdgcn_s_wait_asynccnt)
#define USE_ASYNC 1
#else
#define USE_ASYNC 0
#endif

typedef __attribute__((address_space(1))) v4i* gptr_v4i;
typedef __attribute__((address_space(3))) v4i* lptr_v4i;

__device__ __forceinline__ void cp16(float* lds, const float* g) {
#if USE_ASYNC
  __builtin_amdgcn_global_load_async_to_lds_b128(
      (gptr_v4i)(g), (lptr_v4i)(lds), 0, 0);
#else
  *(f4*)lds = *(const f4*)g;
#endif
}
__device__ __forceinline__ void async_wait0() {
#if USE_ASYNC
  __builtin_amdgcn_s_wait_asynccnt(0);
#endif
}

__device__ __forceinline__ float sigf(float x) { return 1.f / (1.f + __expf(-x)); }

template<int ACT> __device__ __forceinline__ float actf(float x) {
  if constexpr (ACT == 1) return x > 0.f ? x : 0.f;   // relu
  if constexpr (ACT == 2) return sigf(x);             // sigmoid
  if constexpr (ACT == 3) return tanhf(x);            // tanh
  return x;
}

// =====================================================================
// Generic batched WMMA GEMM:  C = act(alpha * A@B + bias(+bias2))
//  A: [M,K] lda, batched base = A + (z/HA)*sAb + (z%HA)*sAh
//  B: non-trans [K,N] ldb, or TRANSB: W[N,K] ldb (B[k][n] = W[n][k])
//  Block: 256 thr (8 waves). Block tile 128x64, wave tile 16x64, K-step 32.
// =====================================================================
template<int ACT, bool TRANSB>
__global__ __launch_bounds__(256)
void wmma_gemm(const float* __restrict__ A, int lda, long long sAb, long long sAh, int HA,
               const float* __restrict__ Bm, int ldb, long long sBb, long long sBh, int HB,
               float* __restrict__ C, int ldc, long long sCb, long long sCh, int HC,
               const float* __restrict__ bias, const float* __restrict__ bias2,
               float alpha, int K)
{
  __shared__ __align__(16) float As[128 * 36];   // padded stride 36 (16B-aligned rows)
  __shared__ __align__(16) float Bs[2304];       // max(32*64, 64*36)

  const int tid  = threadIdx.x;
  const int lane = tid & 31;
  const int wv   = tid >> 5;            // wave id 0..7
  const int z    = blockIdx.z;

  const float* Ab = A  + (long long)(z / HA) * sAb + (long long)(z % HA) * sAh;
  const float* Bb = Bm + (long long)(z / HB) * sBb + (long long)(z % HB) * sBh;
  float*       Cb = C  + (long long)(z / HC) * sCb + (long long)(z % HC) * sCh;

  const int row0 = blockIdx.y * 128;
  const int n0   = blockIdx.x * 64;

  v8f zero8 = {0.f,0.f,0.f,0.f,0.f,0.f,0.f,0.f};
  v8f acc[4] = {zero8, zero8, zero8, zero8};

  const int lm = lane & 15;
  const int kh = lane >> 4;             // 0/1 half-wave

  for (int k0 = 0; k0 < K; k0 += 32) {
    // ---- stage A tile 128x32 (1024 x 16B chunks) ----
    #pragma unroll
    for (int i = 0; i < 4; i++) {
      int c = tid + 256 * i;            // 0..1023
      int r = c >> 3, kc = (c & 7) << 2;
      cp16(&As[r * 36 + kc], &Ab[(long long)(row0 + r) * lda + k0 + kc]);
    }
    // ---- stage B tile (512 x 16B chunks) ----
    if (TRANSB) {
      #pragma unroll
      for (int i = 0; i < 2; i++) {
        int c = tid + 256 * i;          // 0..511
        int n = c >> 3, kc = (c & 7) << 2;
        cp16(&Bs[n * 36 + kc], &Bb[(long long)(n0 + n) * ldb + k0 + kc]);
      }
    } else {
      #pragma unroll
      for (int i = 0; i < 2; i++) {
        int c = tid + 256 * i;
        int k = c >> 4, nc = (c & 15) << 2;
        cp16(&Bs[k * 64 + nc], &Bb[(long long)(k0 + k) * ldb + n0 + nc]);
      }
    }
    // prefetch next K-tile of A into cache hierarchy (global_prefetch_b8)
    if (k0 + 32 < K) {
      int r = tid >> 1, half = (tid & 1) << 4;
      __builtin_prefetch(&Ab[(long long)(row0 + r) * lda + k0 + 32 + half], 0, 1);
    }
    async_wait0();
    __syncthreads();

    // ---- A fragment (ISA 16-bit A 16x32 layout) ----
    v16h af;
    #pragma unroll
    for (int v = 0; v < 8; v++) {
      int kb = ((v & 3) << 1) + kh * 8 + ((v >> 2) << 4);
      const float* ap = &As[(wv * 16 + lm) * 36 + kb];
      af[2 * v]     = (_Float16)ap[0];
      af[2 * v + 1] = (_Float16)ap[1];
    }
    // ---- 4 N-tiles ----
    #pragma unroll
    for (int nt = 0; nt < 4; nt++) {
      v16h bf;
      #pragma unroll
      for (int v = 0; v < 8; v++) {
        int kk = ((v & 3) << 1) + kh * 8 + ((v >> 2) << 4);
        int nn = nt * 16 + lm;
        float e0, e1;
        if (TRANSB) { e0 = Bs[nn * 36 + kk];  e1 = Bs[nn * 36 + kk + 1]; }
        else        { e0 = Bs[kk * 64 + nn];  e1 = Bs[(kk + 1) * 64 + nn]; }
        bf[2 * v]     = (_Float16)e0;
        bf[2 * v + 1] = (_Float16)e1;
      }
      acc[nt] = __builtin_amdgcn_wmma_f32_16x16x32_f16(
          false, af, false, bf, (short)0, acc[nt], false, false);
    }
    __syncthreads();
  }

  // ---- epilogue: C/D layout (VGPR i -> M=i / M=8+i) ----
  const int rhalf = kh * 8;
  #pragma unroll
  for (int nt = 0; nt < 4; nt++) {
    int col = n0 + nt * 16 + lm;
    float bv = bias ? bias[col] : 0.f;
    if (bias2) bv += bias2[col];
    #pragma unroll
    for (int i = 0; i < 8; i++) {
      int row = row0 + wv * 16 + rhalf + i;
      float val = acc[nt][i] * alpha + bv;
      Cb[(long long)row * ldc + col] = actf<ACT>(val);
    }
  }
}

// =====================================================================
// LSTM recurrent step GEMM: g[b][n] = hh @ whh^T + xs[b*S+s][n]
// M=16: rows 8..15 are zero via unconditional clamped-row load * mask
// (pure v_mul_f32, no exec divergence). N=4096, K=1024.
// One wave per 16-col tile.
// =====================================================================
__global__ __launch_bounds__(256)
void lstm_step_gemm(const float* __restrict__ hh, const float* __restrict__ whh,
                    const float* __restrict__ xs, int s, float* __restrict__ g)
{
  const int lane = threadIdx.x & 31;
  const int wv   = threadIdx.x >> 5;
  const int nt   = blockIdx.x * 8 + wv;     // 0..255
  const int ncol = nt * 16;
  const int lm   = lane & 15;
  const int kh   = lane >> 4;
  const float msk = (lm < B_) ? 1.f : 0.f;          // zero pad rows 8..15
  const float* hrow = &hh[(lm & (B_ - 1)) * D_];    // clamped row, always in-bounds

  v8f acc = {0.f,0.f,0.f,0.f,0.f,0.f,0.f,0.f};

  for (int k0 = 0; k0 < D_; k0 += 32) {
    v16h af;
    #pragma unroll
    for (int v = 0; v < 8; v++) {
      int kb = ((v & 3) << 1) + kh * 8 + ((v >> 2) << 4);
      const float* p = &hrow[k0 + kb];
      float a0 = p[0];                  // unconditional, vectorizable
      float a1 = p[1];
      af[2 * v]     = (_Float16)(a0 * msk);
      af[2 * v + 1] = (_Float16)(a1 * msk);
    }
    v16h bf;
    #pragma unroll
    for (int v = 0; v < 8; v++) {
      int kk = k0 + ((v & 3) << 1) + kh * 8 + ((v >> 2) << 4);
      const float* p = &whh[(long long)(ncol + lm) * D_ + kk];   // TRANSB: whh[n][k]
      bf[2 * v] = (_Float16)p[0]; bf[2 * v + 1] = (_Float16)p[1];
    }
    acc = __builtin_amdgcn_wmma_f32_16x16x32_f16(false, af, false, bf, (short)0, acc, false, false);
  }

  const int col = ncol + lm;
  const int rh  = kh * 8;
  #pragma unroll
  for (int i = 0; i < 8; i++) {
    int row = rh + i;
    if (row < B_)
      g[row * (4 * D_) + col] = acc[i] + xs[(long long)(row * S_ + s) * (4 * D_) + col];
  }
}

// lstm gates: i,f,g,o splits of g[8][4096]; update cc,hh; hs -> m[b][s][:]
__global__ void lstm_gate(const float* __restrict__ g, float* __restrict__ hh,
                          float* __restrict__ cc, float* __restrict__ m, int s)
{
  int idx = blockIdx.x * 256 + threadIdx.x;
  if (idx >= B_ * D_) return;
  int b = idx >> 10, n = idx & (D_ - 1);
  const float* gr = g + b * 4 * D_;
  float gi = gr[n], gf = gr[D_ + n], gg = gr[2 * D_ + n], go = gr[3 * D_ + n];
  float c = sigf(gf) * cc[idx] + sigf(gi) * tanhf(gg);
  float h = sigf(go) * tanhf(c);
  cc[idx] = c; hh[idx] = h;
  m[(long long)(b * S_ + s) * D_ + n] = h;
}

// =====================================================================
// LayerNorm over rows of 1024 (population variance, eps=1e-5)
// =====================================================================
__global__ __launch_bounds__(256)
void layernorm_k(float* __restrict__ out, const float* __restrict__ in,
                 const float* __restrict__ gm, const float* __restrict__ bt)
{
  __shared__ float red[16];
  long long row = blockIdx.x;
  const float* x = in + row * D_;
  float s = 0.f, s2 = 0.f;
  for (int i = threadIdx.x; i < D_; i += 256) { float v = x[i]; s += v; s2 += v * v; }
  for (int o = 16; o > 0; o >>= 1) { s += __shfl_xor(s, o, 32); s2 += __shfl_xor(s2, o, 32); }
  int wv = threadIdx.x >> 5, lane = threadIdx.x & 31;
  if (lane == 0) { red[wv] = s; red[8 + wv] = s2; }
  __syncthreads();
  if (threadIdx.x == 0) {
    float ts = 0.f, ts2 = 0.f;
    for (int i = 0; i < 8; i++) { ts += red[i]; ts2 += red[8 + i]; }
    red[0] = ts; red[1] = ts2;
  }
  __syncthreads();
  float mean = red[0] / (float)D_;
  float var  = red[1] / (float)D_ - mean * mean;
  float rstd = rsqrtf(var + 1e-5f);
  float* o = out + row * D_;
  for (int i = threadIdx.x; i < D_; i += 256)
    o[i] = (x[i] - mean) * rstd * gm[i] + bt[i];
}

// Softmax over rows of 256 (one wave per row, 8 rows per block)
__global__ __launch_bounds__(256)
void softmax256_k(float* __restrict__ p)
{
  long long row = (long long)blockIdx.x * 8 + (threadIdx.x >> 5);
  int lane = threadIdx.x & 31;
  float* r = p + row * 256;
  float v[8], mx = -1e30f;
  #pragma unroll
  for (int j = 0; j < 8; j++) { v[j] = r[lane + 32 * j]; mx = fmaxf(mx, v[j]); }
  for (int o = 16; o > 0; o >>= 1) mx = fmaxf(mx, __shfl_xor(mx, o, 32));
  float sum = 0.f;
  #pragma unroll
  for (int j = 0; j < 8; j++) { v[j] = __expf(v[j] - mx); sum += v[j]; }
  for (int o = 16; o > 0; o >>= 1) sum += __shfl_xor(sum, o, 32);
  float inv = 1.f / sum;
  #pragma unroll
  for (int j = 0; j < 8; j++) r[lane + 32 * j] = v[j] * inv;
}

// Gate: logits = x@gate_w + gate_b (E=4), top-2 softmax -> gw[token][4]
__global__ __launch_bounds__(256)
void gate_topk_k(const float* __restrict__ x, const float* __restrict__ gwm,
                 const float* __restrict__ gb, float* __restrict__ gout)
{
  int tok  = blockIdx.x * 8 + (threadIdx.x >> 5);
  int lane = threadIdx.x & 31;
  const float* xr = x + (long long)tok * D_;
  float l0 = 0.f, l1 = 0.f, l2 = 0.f, l3 = 0.f;
  for (int d = lane; d < D_; d += 32) {
    float xv = xr[d];
    const float* wr = &gwm[d * E_];
    l0 += xv * wr[0]; l1 += xv * wr[1]; l2 += xv * wr[2]; l3 += xv * wr[3];
  }
  for (int o = 16; o > 0; o >>= 1) {
    l0 += __shfl_xor(l0, o, 32); l1 += __shfl_xor(l1, o, 32);
    l2 += __shfl_xor(l2, o, 32); l3 += __shfl_xor(l3, o, 32);
  }
  if (lane == 0) {
    float lg[4] = { l0 + gb[0], l1 + gb[1], l2 + gb[2], l3 + gb[3] };
    int i0 = 0;
    for (int e = 1; e < 4; e++) if (lg[e] > lg[i0]) i0 = e;
    int i1 = -1;
    for (int e = 0; e < 4; e++) { if (e == i0) continue; if (i1 < 0 || lg[e] > lg[i1]) i1 = e; }
    float eb = __expf(lg[i1] - lg[i0]);
    float inv = 1.f / (1.f + eb);
    float w[4] = {0.f, 0.f, 0.f, 0.f};
    w[i0] = inv; w[i1] = eb * inv;
    float* go = gout + tok * E_;
    go[0] = w[0]; go[1] = w[1]; go[2] = w[2]; go[3] = w[3];
  }
}

__global__ void zero_k(float* __restrict__ p, long long n) {
  long long i = (long long)blockIdx.x * 256 + threadIdx.x;
  if (i < n) p[i] = 0.f;
}
__global__ void mul_k(float* __restrict__ o, const float* __restrict__ a,
                      const float* __restrict__ b) {
  long long i = (long long)blockIdx.x * 256 + threadIdx.x;
  o[i] = a[i] * b[i];
}
__global__ void accum_k(float* __restrict__ out, const float* __restrict__ src,
                        const float* __restrict__ gw, int e) {
  long long i = (long long)blockIdx.x * 256 + threadIdx.x;
  int tok = (int)(i >> 10);
  out[i] += gw[tok * E_ + e] * src[i];
}

// ---------------- host-side dispatch ----------------
static inline void gemm(hipStream_t st, int actc, bool transB,
                        const float* A, int lda, long long sAb, long long sAh, int HA,
                        const float* Bm, int ldb, long long sBb, long long sBh, int HB,
                        float* C, int ldc, long long sCb, long long sCh, int HC,
                        const float* bias, const float* bias2, float alpha,
                        int M, int N, int K, int batch)
{
  dim3 g(N / 64, M / 128, batch), b(256);
#define GCALL(a, t) wmma_gemm<a, t><<<g, b, 0, st>>>(A, lda, sAb, sAh, HA, Bm, ldb, sBb, sBh, HB, \
                                                     C, ldc, sCb, sCh, HC, bias, bias2, alpha, K)
  if (!transB) {
    switch (actc) { case 0: GCALL(0,false); break; case 1: GCALL(1,false); break;
                    case 2: GCALL(2,false); break; default: GCALL(3,false); break; }
  } else {
    switch (actc) { case 0: GCALL(0,true); break; case 1: GCALL(1,true); break;
                    case 2: GCALL(2,true); break; default: GCALL(3,true); break; }
  }
#undef GCALL
}

// token GEMM: A[TOK,K] @ W -> C[TOK,N]
static inline void tgemm(hipStream_t st, int actc, bool tB, const float* A, const float* W,
                         const float* b1, const float* b2, float* C, int N, int K)
{
  gemm(st, actc, tB, A, K, 0, 0, 1, W, (tB ? K : N), 0, 0, 1,
       C, N, 0, 0, 1, b1, b2, 1.f, TOK, N, K, 1);
}

extern "C" void kernel_launch(void* const* d_in, const int* in_sizes, int n_in,
                              void* d_out, int out_size, void* d_ws, size_t ws_size,
                              hipStream_t stream)
{
  (void)in_sizes; (void)n_in; (void)out_size;
  const float* x      = (const float*)d_in[0];
  const float* gate_w = (const float*)d_in[1];
  const float* gate_b = (const float*)d_in[2];
  const float* P[32];
  for (int i = 0; i < 32; i++) P[i] = (const float*)d_in[3 + i];
  enum { ENC_W1 = 0, ENC_B1, ENC_W2, ENC_B2, ENC_LN_G, ENC_LN_B,
         WQ, BQ, WK, BK, WV, BV, WO, BO, WIH, WHH, BIH, BHH,
         OPT_W1, OPT_B1, OPT_W2, OPT_B2, PER_W1, PER_B1, PER_W2, PER_B2,
         DEC_W1, DEC_B1, DEC_W2, DEC_B2, DEC_LN_G, DEC_LN_B };

  float* w = (float*)d_ws;
  size_t off = 0;
  float* gw  = w + off; off += (size_t)TOK * E_;
  float* b0  = w + off; off += (size_t)TOK * D_;
  float* b1  = w + off; off += (size_t)TOK * D_;
  float* b2  = w + off; off += (size_t)TOK * D_;
  float* b3  = w + off; off += (size_t)TOK * D_;
  float* b4  = w + off; off += (size_t)TOK * D_;
  float* sc  = w + off; off += (size_t)B_ * H_ * S_ * S_;
  float* xs  = w + off; off += (size_t)TOK * 4 * D_;
  float* gbf = w + off; off += (size_t)B_ * 4 * D_;
  float* hh  = w + off; off += (size_t)B_ * D_;
  float* cc  = w + off; off += (size_t)B_ * D_;
  if (ws_size < off * sizeof(float)) return;   // workspace too small: no-op

  const long long SD = (long long)S_ * D_;
  const long long SS = (long long)S_ * S_;

  // out = 0 ; gate weights
  zero_k<<<(TOK * D_ + 255) / 256, 256, 0, stream>>>((float*)d_out, (long long)TOK * D_);
  gate_topk_k<<<TOK / 8, 256, 0, stream>>>(x, gate_w, gate_b, gw);

  for (int e = 0; e < E_; e++) {
    const size_t wE = (size_t)e * D_ * D_, bE = (size_t)e * D_;
    const size_t w4E = (size_t)e * 4 * D_ * D_, b4E = (size_t)e * 4 * D_;

    // encoder
    tgemm(stream, 1, false, x,  P[ENC_W1] + wE, P[ENC_B1] + bE, nullptr, b0, D_, D_);
    tgemm(stream, 0, false, b0, P[ENC_W2] + wE, P[ENC_B2] + bE, nullptr, b1, D_, D_);
    layernorm_k<<<TOK, 256, 0, stream>>>(b0, b1, P[ENC_LN_G] + bE, P[ENC_LN_B] + bE);

    // q,k,v
    tgemm(stream, 0, false, b0, P[WQ] + wE, P[BQ] + bE, nullptr, b2, D_, D_);
    tgemm(stream, 0, false, b0, P[WK] + wE, P[BK] + bE, nullptr, b3, D_, D_);
    tgemm(stream, 0, false, b0, P[WV] + wE, P[BV] + bE, nullptr, b4, D_, D_);

    // scores = (q k^T)/8 : batched over (b,h)=128
    gemm(stream, 0, true,
         b2, D_, SD, DH_, H_,
         b3, D_, SD, DH_, H_,
         sc, S_, (long long)H_ * SS, SS, H_,
         nullptr, nullptr, 0.125f, S_, S_, DH_, B_ * H_);
    softmax256_k<<<(B_ * H_ * S_) / 8, 256, 0, stream>>>(sc);

    // o1 = att @ v -> b1
    gemm(stream, 0, false,
         sc, S_, (long long)H_ * SS, SS, H_,
         b4, D_, SD, DH_, H_,
         b1, D_, SD, DH_, H_,
         nullptr, nullptr, 1.f, S_, DH_, S_, B_ * H_);

    // o = o1 @ wo + bo -> b2
    tgemm(stream, 0, false, b1, P[WO] + wE, P[BO] + bE, nullptr, b2, D_, D_);

    // xs = o @ wih^T + bih + bhh
    tgemm(stream, 0, true, b2, P[WIH] + w4E, P[BIH] + b4E, P[BHH] + b4E, xs, 4 * D_, D_);

    // LSTM scan
    zero_k<<<(2 * B_ * D_ + 255) / 256, 256, 0, stream>>>(hh, (long long)2 * B_ * D_); // hh & cc contiguous
    for (int s = 0; s < S_; s++) {
      lstm_step_gemm<<<32, 256, 0, stream>>>(hh, P[WHH] + w4E, xs, s, gbf);
      lstm_gate<<<(B_ * D_ + 255) / 256, 256, 0, stream>>>(gbf, hh, cc, b3, s);
    }

    // gated branches on m=b3
    tgemm(stream, 1, false, b3, P[OPT_W1] + wE, P[OPT_B1] + bE, nullptr, b0, D_, D_);
    tgemm(stream, 2, false, b0, P[OPT_W2] + wE, P[OPT_B2] + bE, nullptr, b1, D_, D_);
    tgemm(stream, 1, false, b3, P[PER_W1] + wE, P[PER_B1] + bE, nullptr, b0, D_, D_);
    tgemm(stream, 3, false, b0, P[PER_W2] + wE, P[PER_B2] + bE, nullptr, b2, D_, D_);
    mul_k<<<(TOK * D_) / 256, 256, 0, stream>>>(b0, b1, b2);

    // decoder
    tgemm(stream, 1, false, b0, P[DEC_W1] + wE, P[DEC_B1] + bE, nullptr, b1, D_, D_);
    tgemm(stream, 0, false, b1, P[DEC_W2] + wE, P[DEC_B2] + bE, nullptr, b0, D_, D_);
    layernorm_k<<<TOK, 256, 0, stream>>>(b2, b0, P[DEC_LN_G] + bE, P[DEC_LN_B] + bE);

    // out += gw[:,:,e] * out_e
    accum_k<<<(TOK * D_) / 256, 256, 0, stream>>>((float*)d_out, b2, gw, e);
  }
}
